// CrossAttention_49615462203535
// MI455X (gfx1250) — matrix operand order repaired and checked by reading, not compiled, not run on hardware
//
#include <hip/hip_runtime.h>
#include <stdint.h>

typedef __attribute__((ext_vector_type(16))) __bf16 v16bf;
typedef __attribute__((ext_vector_type(8)))  float  v8f;
typedef unsigned short u16;
typedef unsigned int   u32;

#define EMBED 1024
#define HEADS 16
#define HDIM  64
#define BATCH 4
#define SEQ   2048
#define KTC   32          // 1024 / 32 : number of 32-wide k-steps
#define FRAG  512         // elems per 16x32 / 32x16 fragment (32 lanes x 16 bf16)
#define TILE  4096        // elems per 8-fragment k-step tile (8 KB)

// ---------------------------------------------------------------------------
// helpers
// ---------------------------------------------------------------------------
__device__ __forceinline__ u16 f2bf(float f) {
  union { float f; u32 u; } v; v.f = f;
  u32 r = v.u + 0x7FFFu + ((v.u >> 16) & 1u);   // round-to-nearest-even
  return (u16)(r >> 16);
}
__device__ __forceinline__ u32 f2bf2(float lo, float hi) {
  return (u32)f2bf(lo) | ((u32)f2bf(hi) << 16);
}
__device__ __forceinline__ v8f zero8() {
  v8f v;
#pragma unroll
  for (int i = 0; i < 8; ++i) v[i] = 0.0f;
  return v;
}

// LDS byte-offset of a generic pointer to a __shared__ object (addrspace(3))
typedef __attribute__((address_space(3))) unsigned short lds_u16;
__device__ __forceinline__ u32 as_lds(const u16* p) {
  return (u32)(size_t)(const lds_u16*)p;
}

// CDNA5 async DMA: 16 bytes per lane, global -> LDS, tracked by ASYNCcnt
__device__ __forceinline__ void async_load_b128(u32 lds_off, const void* gaddr) {
  asm volatile("global_load_async_to_lds_b128 %0, %1, off"
               :: "v"(lds_off), "v"(gaddr) : "memory");
}
__device__ __forceinline__ void wait_async0() {
  asm volatile("s_wait_asynccnt 0" ::: "memory");
}
__device__ __forceinline__ void wait_ds0() {
  asm volatile("s_wait_dscnt 0" ::: "memory");
}

// 16-lane butterfly reductions in pure VALU via DPP (no LDS traffic).
// xor1 = quad_perm(1,0,3,2); xor2 = quad_perm(2,3,0,1);
// row_half_mirror acts as xor4 once quads are uniform; row_mirror as xor8.
template <int CTRL>
__device__ __forceinline__ float dpp_fmax(float x) {
  union { float f; int i; } a, b; a.f = x;
  b.i = __builtin_amdgcn_update_dpp(0, a.i, CTRL, 0xf, 0xf, true);
  return fmaxf(x, b.f);
}
template <int CTRL>
__device__ __forceinline__ float dpp_fadd(float x) {
  union { float f; int i; } a, b; a.f = x;
  b.i = __builtin_amdgcn_update_dpp(0, a.i, CTRL, 0xf, 0xf, true);
  return x + b.f;
}
__device__ __forceinline__ float red16_max(float x) {
  x = dpp_fmax<0xB1>(x);   // xor1
  x = dpp_fmax<0x4E>(x);   // xor2
  x = dpp_fmax<0x141>(x);  // row_half_mirror (xor4 given quad-uniform)
  x = dpp_fmax<0x140>(x);  // row_mirror      (xor8 given 8-uniform)
  return x;
}
__device__ __forceinline__ float red16_sum(float x) {
  x = dpp_fadd<0xB1>(x);
  x = dpp_fadd<0x4E>(x);
  x = dpp_fadd<0x141>(x);
  x = dpp_fadd<0x140>(x);
  return x;
}

// ---------------------------------------------------------------------------
// Fragment-order global layouts (so hot loops are pure async-DMA + WMMA).
//  A 16x32 frag: lane = (m&15) + 16*(kk>>4),     e = kk&15
//  B 32x16 frag: lane = (n&15) + 16*(kk>>4),     e = kk&15
// A tiled by 128-row m-blocks; B tiled by 256-col n-blocks (16 frags / k-step)
// ---------------------------------------------------------------------------
__device__ __forceinline__ size_t afrag_index(int m, int k) {
  const int mblock = m >> 7, mf = (m >> 4) & 7, r = m & 15;
  const int kt = k >> 5, kk = k & 31;
  const int lanep = r + 16 * (kk >> 4);
  const int e = kk & 15;
  return ((((size_t)mblock * KTC + kt) * 8 + mf) * 32 + lanep) * 16 + e;
}
__device__ __forceinline__ size_t bfrag_index(int k, int n) {
  const int nblock = n >> 8, nf = (n >> 4) & 15, cc = n & 15;
  const int kt = k >> 5, kk = k & 31;
  const int lanep = cc + 16 * (kk >> 4);
  const int e = kk & 15;
  return ((((size_t)nblock * KTC + kt) * 16 + nf) * 32 + lanep) * 16 + e;
}
// Attention K^T tile layout: per (bh, ktile64): frag(t=d>>5, nf=kpos>>4)
__device__ __forceinline__ size_t attnK_index(int gm, int gn) {
  const int b = gm >> 11, s = gm & (SEQ - 1);
  const int h = gn >> 6, d = gn & 63;
  const int bh = b * HEADS + h;
  const int ktile = s >> 6, kpos = s & 63;
  const int frag = (d >> 5) * 4 + (kpos >> 4);
  const int lanep = (kpos & 15) + 16 * ((d >> 4) & 1);
  const int e = d & 15;
  return ((((size_t)bh * (SEQ / 64) + ktile) * 8 + frag) * 32 + lanep) * 16 + e;
}
// Attention V tile layout: per (bh, ktile64): frag(t=kpos>>5, nf=d>>4)
__device__ __forceinline__ size_t attnV_index(int gm, int gn) {
  const int b = gm >> 11, s = gm & (SEQ - 1);
  const int h = gn >> 6, d = gn & 63;
  const int bh = b * HEADS + h;
  const int ktile = s >> 6, kpos = s & 63;
  const int kk = kpos & 31;
  const int frag = (kpos >> 5) * 4 + (d >> 4);
  const int lanep = (d & 15) + 16 * (kk >> 4);
  const int e = kk & 15;
  return ((((size_t)bh * (SEQ / 64) + ktile) * 8 + frag) * 32 + lanep) * 16 + e;
}

// ---------------------------------------------------------------------------
// Pre-pass: fp32 row-major activations [8192 x 1024] -> bf16 A-fragment layout
// ---------------------------------------------------------------------------
__global__ __launch_bounds__(256) void convert_a_afrag(
    const float* __restrict__ A, u16* __restrict__ out)
{
  const int t = blockIdx.x * 256 + threadIdx.x;   // 2M chunks of 4
  const int m = t >> 8;
  const int k = (t & 255) * 4;
  const float4 f = *(const float4*)&A[(size_t)m * EMBED + k];
  uint2 pk; pk.x = f2bf2(f.x, f.y); pk.y = f2bf2(f.z, f.w);
  *(uint2*)&out[afrag_index(m, k)] = pk;          // 4 consecutive e slots
}

// Pre-pass: fp32 weights [1024 x 1024] -> bf16 B-fragment layout
__global__ __launch_bounds__(256) void convert_w_bfrag(
    const float* __restrict__ W, u16* __restrict__ out)
{
  const int t = blockIdx.x * 256 + threadIdx.x;   // 256K chunks of 4
  const int n = t & (EMBED - 1);
  const int k = (t >> 10) * 4;
  const float x0 = W[(size_t)k * EMBED + n];
  const float x1 = W[(size_t)(k + 1) * EMBED + n];
  const float x2 = W[(size_t)(k + 2) * EMBED + n];
  const float x3 = W[(size_t)(k + 3) * EMBED + n];
  uint2 pk; pk.x = f2bf2(x0, x1); pk.y = f2bf2(x2, x3);
  *(uint2*)&out[bfrag_index(k, n)] = pk;          // 4 consecutive e slots
}

// ---------------------------------------------------------------------------
// GEMM: C[M,N] = A*B + bias. A/B pre-swizzled bf16 fragment tiles in global.
// Block 256 thr = 8 waves (2x4); block tile 128x256; wave tile 64x64
// (4x4 accumulator frags -> 32 WMMA per 32 ds_load_b128 per 64-deep k-tile).
// Async-DMA double-buffered: 16KB (A) + 32KB (B) per k-tile.
// OUTMODE: 0=f32 row-major, 1=bf16 row-major, 2=attn-K layout, 3=attn-V layout
// ---------------------------------------------------------------------------
template <int OUTMODE>
__global__ __launch_bounds__(256) void gemm_frag_wmma(
    const u16* __restrict__ Af, const u16* __restrict__ Bf,
    const float* __restrict__ bias, void* __restrict__ Cout,
    int M, int N, int K)
{
  __shared__ __align__(32) u16 lds_a[2][2 * TILE];       // 2 x 16KB
  __shared__ __align__(32) u16 lds_b[2][2 * 2 * TILE];   // 2 x 32KB

  const int tid  = threadIdx.x;
  const int lane = tid & 31;
  const int wave = tid >> 5;
  const int wm = wave >> 2, wn = wave & 3;   // 2 x 4 wave grid
  const int hi = lane >> 4, c = lane & 15;
  const int NT2 = K >> 6;                    // 64-deep k-tiles

  const u16* atile = Af + (size_t)blockIdx.x * (K >> 5) * TILE;
  const u16* btile = Bf + (size_t)blockIdx.y * (K >> 5) * (2 * TILE);

  auto issue = [&](int kt2, int buf) {
    const u16* ga = atile + (size_t)kt2 * (2 * TILE);
    const u16* gb = btile + (size_t)kt2 * (4 * TILE);
#pragma unroll
    for (int j = 0; j < 4; ++j) {
      const int eo = (j * 256 + tid) * 8;               // 16B per lane
      async_load_b128(as_lds(&lds_a[buf][eo]), ga + eo);
    }
#pragma unroll
    for (int j = 0; j < 8; ++j) {
      const int eo = (j * 256 + tid) * 8;
      async_load_b128(as_lds(&lds_b[buf][eo]), gb + eo);
    }
  };

  v8f acc[4][4];
#pragma unroll
  for (int i = 0; i < 4; ++i)
#pragma unroll
    for (int j = 0; j < 4; ++j) acc[i][j] = zero8();

  issue(0, 0);
  for (int kt2 = 0; kt2 < NT2; ++kt2) {
    const int cur = kt2 & 1;
    wait_async0();                 // this thread's DMA for tile kt2 done
    __syncthreads();               // everyone's DMA done; prev buffer free
    if (kt2 + 1 < NT2) issue(kt2 + 1, cur ^ 1);
#pragma unroll
    for (int ks = 0; ks < 2; ++ks) {
      v16bf a[4];
#pragma unroll
      for (int mf = 0; mf < 4; ++mf)
        a[mf] = *(const v16bf*)&lds_a[cur][ks * TILE + ((wm * 4 + mf) * 32 + lane) * 16];
#pragma unroll
      for (int nf = 0; nf < 4; ++nf) {
        v16bf b = *(const v16bf*)&lds_b[cur][ks * (2 * TILE) + ((wn * 4 + nf) * 32 + lane) * 16];
#pragma unroll
        for (int mf = 0; mf < 4; ++mf) {
          acc[mf][nf] = __builtin_amdgcn_wmma_f32_16x16x32_bf16(
              false, a[mf], false, b, (short)0, acc[mf][nf], false, false);
        }
      }
    }
  }

  // writeback (scatter; swizzle happens here, once, outside hot loops)
#pragma unroll
  for (int mf = 0; mf < 4; ++mf) {
#pragma unroll
    for (int nf = 0; nf < 4; ++nf) {
      const int gn = blockIdx.y * 256 + wn * 64 + nf * 16 + c;
      const float bs = bias[gn];
#pragma unroll
      for (int r = 0; r < 8; ++r) {
        const int gm = blockIdx.x * 128 + wm * 64 + mf * 16 + r + 8 * hi;
        const float v = acc[mf][nf][r] + bs;
        if (OUTMODE == 0)      ((float*)Cout)[(size_t)gm * N + gn] = v;
        else if (OUTMODE == 1) ((u16*)Cout)[(size_t)gm * N + gn] = f2bf(v);
        else if (OUTMODE == 2) ((u16*)Cout)[attnK_index(gm, gn)] = f2bf(v);
        else                   ((u16*)Cout)[attnV_index(gm, gn)] = f2bf(v);
      }
    }
  }
}

// ---------------------------------------------------------------------------
// Flash attention per (b,h): K/V tiles arrive pre-fragmented via async DMA,
// double-buffered. Block 128 thr (4 waves); wave owns 16 q-rows.
// Softmax row reductions are DPP butterflies (no LDS).
// Output written directly in A-fragment layout for the final projection GEMM.
// ---------------------------------------------------------------------------
__global__ __launch_bounds__(128) void attn_fa_wmma(
    const u16* __restrict__ Qp, const u16* __restrict__ Kf,
    const u16* __restrict__ Vf, u16* __restrict__ Ap)
{
  __shared__ __align__(32) u16 lds_k[2][TILE];
  __shared__ __align__(32) u16 lds_v[2][TILE];
  __shared__ __align__(32) u16 lds_p[4 * 2 * FRAG];   // per-wave P staging

  const int tid  = threadIdx.x;
  const int lane = tid & 31;
  const int wave = tid >> 5;
  const int hi = lane >> 4, c = lane & 15;

  const int q0 = blockIdx.x * 64;
  const int bh = blockIdx.y;
  const int b  = bh / HEADS;
  const int h  = bh % HEADS;
  const int hd = h * HDIM;

  auto issue_kv = [&](int kt, int buf) {
    const size_t base = ((size_t)bh * (SEQ / 64) + kt) * TILE;
#pragma unroll
    for (int j = 0; j < 4; ++j) {
      const int eo = (j * 128 + tid) * 8;
      async_load_b128(as_lds(&lds_k[buf][eo]), Kf + base + eo);
      async_load_b128(as_lds(&lds_v[buf][eo]), Vf + base + eo);
    }
  };

  // this wave's Q as two A-fragments straight from row-major global
  v16bf a_q[2];
  {
    const u16* qb = Qp + (size_t)(b * SEQ + q0 + wave * 16 + c) * EMBED + hd;
#pragma unroll
    for (int t = 0; t < 2; ++t) {
      union { v16bf v; uint4 q[2]; } u;
      u.q[0] = *(const uint4*)(qb + t * 32 + 16 * hi);
      u.q[1] = *(const uint4*)(qb + t * 32 + 16 * hi + 8);
      a_q[t] = u.v;
    }
  }

  float mstate[8], lstate[8];
  v8f oacc[4];
#pragma unroll
  for (int r = 0; r < 8; ++r) { mstate[r] = -1e30f; lstate[r] = 0.0f; }
#pragma unroll
  for (int nf = 0; nf < 4; ++nf) oacc[nf] = zero8();

  const float SL2E = 0.18033688011112042f;   // log2(e) / sqrt(64)

  issue_kv(0, 0);
  for (int kt = 0; kt < SEQ / 64; ++kt) {
    const int cur = kt & 1;
    wait_async0();
    __syncthreads();
    if (kt + 1 < SEQ / 64) issue_kv(kt + 1, cur ^ 1);

    // ---- S = Q K^T ----
    v8f s[4];
#pragma unroll
    for (int nf = 0; nf < 4; ++nf) {
      s[nf] = zero8();
#pragma unroll
      for (int t = 0; t < 2; ++t) {
        v16bf bk = *(const v16bf*)&lds_k[cur][((t * 4 + nf) * 32 + lane) * 16];
        s[nf] = __builtin_amdgcn_wmma_f32_16x16x32_bf16(
            false, a_q[t], false, bk, (short)0, s[nf], false, false);
      }
    }

    // ---- online softmax (row = r + 8*hi), DPP reductions ----
#pragma unroll
    for (int r = 0; r < 8; ++r) {
      float mx = fmaxf(fmaxf(s[0][r], s[1][r]), fmaxf(s[2][r], s[3][r]));
      mx = red16_max(mx);
      const float mnew  = fmaxf(mstate[r], mx);
      const float alpha = exp2f((mstate[r] - mnew) * SL2E);
      float rowsum = 0.0f;
#pragma unroll
      for (int nf = 0; nf < 4; ++nf) {
        const float p = exp2f((s[nf][r] - mnew) * SL2E);
        rowsum += p;
        const int hip   = nf & 1;                   // kk = (nf&1)*16 + c
        const int lanep = (r + 8 * hi) + 16 * hip;
        const int e     = c;
        lds_p[((wave * 2 + (nf >> 1)) * 32 + lanep) * 16 + e] = f2bf(p);
      }
      rowsum = red16_sum(rowsum);
      lstate[r] = lstate[r] * alpha + rowsum;
      mstate[r] = mnew;
#pragma unroll
      for (int nf = 0; nf < 4; ++nf) oacc[nf][r] *= alpha;
    }

    wait_ds0();   // intra-wave fence for the cross-lane P bounce

    // ---- O += P V ----
#pragma unroll
    for (int t = 0; t < 2; ++t) {
      v16bf ap = *(const v16bf*)&lds_p[((wave * 2 + t) * 32 + lane) * 16];
#pragma unroll
      for (int nf = 0; nf < 4; ++nf) {
        v16bf bv = *(const v16bf*)&lds_v[cur][((t * 4 + nf) * 32 + lane) * 16];
        oacc[nf] = __builtin_amdgcn_wmma_f32_16x16x32_bf16(
            false, ap, false, bv, (short)0, oacc[nf], false, false);
      }
    }
  }

  // ---- normalize + write directly into final-GEMM A-fragment layout ----
#pragma unroll
  for (int nf = 0; nf < 4; ++nf) {
#pragma unroll
    for (int r = 0; r < 8; ++r) {
      const int gm  = b * SEQ + q0 + wave * 16 + r + 8 * hi;
      const int col = hd + nf * 16 + c;
      Ap[afrag_index(gm, col)] = f2bf(oacc[nf][r] / lstate[r]);
    }
  }
}

// ---------------------------------------------------------------------------
extern "C" void kernel_launch(void* const* d_in, const int* in_sizes, int n_in,
                              void* d_out, int out_size, void* d_ws, size_t ws_size,
                              hipStream_t stream) {
  (void)in_sizes; (void)n_in; (void)out_size; (void)ws_size;
  const float* query = (const float*)d_in[0];
  const float* key_  = (const float*)d_in[1];
  const float* value = (const float*)d_in[2];
  const float* Wq = (const float*)d_in[3];
  const float* bq = (const float*)d_in[4];
  const float* Wk = (const float*)d_in[5];
  const float* bk = (const float*)d_in[6];
  const float* Wv = (const float*)d_in[7];
  const float* bv = (const float*)d_in[8];
  const float* Wo = (const float*)d_in[9];
  const float* bo = (const float*)d_in[10];

  const int M = BATCH * SEQ;                  // 8192
  const size_t actE = (size_t)M * EMBED;      // 8M elems (16 MB bf16)
  const size_t wE   = (size_t)EMBED * EMBED;  // 1M elems (2 MB bf16)

  u16* qa  = (u16*)d_ws;        // query  in A-frag layout
  u16* ka  = qa  + actE;        // key    in A-frag layout
  u16* va  = ka  + actE;        // value  in A-frag layout
  u16* wqf = va  + actE;        // weights in B-frag layout
  u16* wkf = wqf + wE;
  u16* wvf = wkf + wE;
  u16* wof = wvf + wE;
  u16* Qp  = wof + wE;          // projected Q, row-major bf16
  u16* Kat = Qp  + actE;        // projected K, attn-K frag layout
  u16* Vat = Kat + actE;        // projected V, attn-V frag layout
  u16* Apf = qa;                // attn output (A-frag) — reuses qa (dead by then)

  // pre-swizzle passes (one-time, bandwidth-trivial)
  convert_a_afrag<<<8192, 256, 0, stream>>>(query, qa);
  convert_a_afrag<<<8192, 256, 0, stream>>>(key_,  ka);
  convert_a_afrag<<<8192, 256, 0, stream>>>(value, va);
  convert_w_bfrag<<<1024, 256, 0, stream>>>(Wq, wqf);
  convert_w_bfrag<<<1024, 256, 0, stream>>>(Wk, wkf);
  convert_w_bfrag<<<1024, 256, 0, stream>>>(Wv, wvf);
  convert_w_bfrag<<<1024, 256, 0, stream>>>(Wo, wof);

  dim3 gblk(256), ggrd(M / 128, EMBED / 256);
  gemm_frag_wmma<1><<<ggrd, gblk, 0, stream>>>(qa, wqf, bq, (void*)Qp,  M, EMBED, EMBED);
  gemm_frag_wmma<2><<<ggrd, gblk, 0, stream>>>(ka, wkf, bk, (void*)Kat, M, EMBED, EMBED);
  gemm_frag_wmma<3><<<ggrd, gblk, 0, stream>>>(va, wvf, bv, (void*)Vat, M, EMBED, EMBED);

  dim3 ablk(128), agrd(SEQ / 64, BATCH * HEADS);
  attn_fa_wmma<<<agrd, ablk, 0, stream>>>(Qp, Kat, Vat, Apf);

  gemm_frag_wmma<0><<<ggrd, gblk, 0, stream>>>(Apf, wof, bo, d_out, M, EMBED, EMBED);
}
